// HungarianMatcher_24481313587585
// MI455X (gfx1250) — matrix operand order repaired and checked by reading, compiled, not run on hardware
//
#include <hip/hip_runtime.h>

typedef __attribute__((ext_vector_type(16))) _Float16 v16h;
typedef __attribute__((ext_vector_type(8)))  float    v8f;
typedef __attribute__((ext_vector_type(4)))  float    f32x4;

#define B_   8
#define N_   128
#define M_   32
#define C_   65536
#define NT_  8          // n-tiles of 16
#define CS_  8          // c splits
#define WAVES_ 4
#define COLS_PER_WAVE (C_/(CS_*WAVES_))   // 2048

// workspace layout (float offsets)
#define WS1_OFF 0
#define WS1_SZ  (B_*NT_*CS_*3*512)        // [b][nt][cs][mat=3][16][32]
#define WS2_OFF (WS1_OFF + WS1_SZ)
#define WS2_SZ  (B_*NT_*CS_*32)           // [b][nt][cs][{neg,sig}][16]
#define WS3_OFF (WS2_OFF + WS2_SZ)        // [b][cs][32]

#define LOG2E_F 1.4426950408889634f
#define LN2_F   0.6931471805599453f

__device__ __forceinline__ v8f wmma_f16(v16h a, v16h b, v8f c) {
  return __builtin_amdgcn_wmma_f32_16x16x32_f16(
      /*neg_a=*/false, a, /*neg_b=*/false, b,
      /*c_mod=*/(short)0, c, /*reuse_a=*/false, /*reuse_b=*/false);
}

// ---------------- Phase 1: partial einsums via WMMA -------------------------
__global__ __launch_bounds__(128)
void cost_partial_kernel(const float* __restrict__ pred,
                         const float* __restrict__ gt,
                         float* __restrict__ ws) {
  const int cs  = blockIdx.x % CS_;
  const int nt  = (blockIdx.x / CS_) % NT_;
  const int b   = blockIdx.x / (CS_*NT_);
  const int tid = threadIdx.x;
  const int wave = tid >> 5;
  const int lane = tid & 31;
  const int h = lane >> 4;      // K-half selector (ISA A/B layouts)
  const int r = lane & 15;      // row within tile
  const bool need_gsum = (nt == 0);   // block-uniform: gt row-sums only from nt==0

  const float* prow  = pred + (size_t)(b*N_ + nt*16 + r) * C_;
  const float* grow0 = gt   + (size_t)(b*M_ + r)        * C_;
  const float* grow1 = gt   + (size_t)(b*M_ + 16 + r)   * C_;

  v8f acc[6] = {};   // [mat(P,Ng,S)][mtile]
  float nsum = 0.f, ssum = 0.f, g0s = 0.f, g1s = 0.f;

  const int cbase = cs * (C_/CS_) + wave * COLS_PER_WAVE;
  for (int c0 = cbase; c0 < cbase + COLS_PER_WAVE; c0 += 32) {
    // ---- A fragment: ISA 16-bit A layout: element e -> K = 8h + e (+8 if e>=8)
    const f32x4* pa = (const f32x4*)(prow + c0 + 8*h);
    f32x4 x0 = __builtin_nontemporal_load(pa);
    f32x4 x1 = __builtin_nontemporal_load(pa + 1);
    const f32x4* pb = (const f32x4*)(prow + c0 + 8*h + 16);
    f32x4 x2 = __builtin_nontemporal_load(pb);
    f32x4 x3 = __builtin_nontemporal_load(pb + 1);
    float av[16] = {x0.x,x0.y,x0.z,x0.w, x1.x,x1.y,x1.z,x1.w,
                    x2.x,x2.y,x2.z,x2.w, x3.x,x3.y,x3.z,x3.w};

    v16h a_pos, a_neg, a_sig;
    #pragma unroll
    for (int e = 0; e < 16; e++) {
      float x   = av[e];
      // e^{-|x|} via raw v_exp_f32 (arg <= 0, underflows cleanly)
      float em  = __builtin_amdgcn_exp2f(fabsf(x) * -LOG2E_F);
      float op  = 1.0f + em;                         // in [1, 2]
      // ln(1+em) via raw v_log_f32 (arg in [1,2]: no special cases)
      float l   = LN2_F * __builtin_amdgcn_logf(op);
      float spn = fmaxf(x, 0.f) + l;                 // softplus(x)
      float spp = spn - x;                           // softplus(-x)
      float inv = __builtin_amdgcn_rcpf(op);
      float sig = (x >= 0.f) ? inv : em * inv;       // sigmoid(x)
      a_neg[e] = (_Float16)spn;
      a_pos[e] = (_Float16)spp;
      a_sig[e] = (_Float16)sig;
      nsum += spn; ssum += sig;
    }

    // ---- B fragments: ISA 16-bit B layout: element e -> K = 8h + e (+8 if e>=8)
    const f32x4* q0 = (const f32x4*)(grow0 + c0 + 8*h);
    f32x4 y0 = q0[0], y1 = q0[1], y2 = q0[4], y3 = q0[5];
    const f32x4* q1 = (const f32x4*)(grow1 + c0 + 8*h);
    f32x4 z0 = q1[0], z1 = q1[1], z2 = q1[4], z3 = q1[5];
    float bv0[16] = {y0.x,y0.y,y0.z,y0.w, y1.x,y1.y,y1.z,y1.w,
                     y2.x,y2.y,y2.z,y2.w, y3.x,y3.y,y3.z,y3.w};
    float bv1[16] = {z0.x,z0.y,z0.z,z0.w, z1.x,z1.y,z1.z,z1.w,
                     z2.x,z2.y,z2.z,z2.w, z3.x,z3.y,z3.z,z3.w};
    v16h b0, b1;
    #pragma unroll
    for (int e = 0; e < 16; e++) {
      b0[e] = (_Float16)bv0[e];
      b1[e] = (_Float16)bv1[e];
    }
    if (need_gsum) {            // uniform branch; hot loop skips this in 7/8 blocks
      #pragma unroll
      for (int e = 0; e < 16; e++) { g0s += bv0[e]; g1s += bv1[e]; }
    }

    acc[0] = wmma_f16(a_pos, b0, acc[0]);
    acc[1] = wmma_f16(a_pos, b1, acc[1]);
    acc[2] = wmma_f16(a_neg, b0, acc[2]);
    acc[3] = wmma_f16(a_neg, b1, acc[3]);
    acc[4] = wmma_f16(a_sig, b0, acc[4]);
    acc[5] = wmma_f16(a_sig, b1, acc[5]);
  }

  // ---- cross-wave reduction via LDS ----
  __shared__ float lds_mat[WAVES_][3][16][32];
  __shared__ float lds_ns[WAVES_][2][16];
  __shared__ float lds_gs[WAVES_][32];

  // C/D layout: lane l, vgpr v -> (n_local = v + 8*(l>>4), m_local = l&15)
  #pragma unroll
  for (int q = 0; q < 3; q++)
    #pragma unroll
    for (int t = 0; t < 2; t++)
      #pragma unroll
      for (int v = 0; v < 8; v++)
        lds_mat[wave][q][v + 8*h][t*16 + r] = acc[q*2 + t][v];

  nsum += __shfl_xor(nsum, 16);
  ssum += __shfl_xor(ssum, 16);
  g0s  += __shfl_xor(g0s, 16);
  g1s  += __shfl_xor(g1s, 16);
  if (h == 0) {
    lds_ns[wave][0][r] = nsum;
    lds_ns[wave][1][r] = ssum;
    lds_gs[wave][r]      = g0s;
    lds_gs[wave][16 + r] = g1s;
  }
  __syncthreads();

  const size_t base1 = (size_t)((b*NT_ + nt)*CS_ + cs) * 3 * 512;
  for (int idx = tid; idx < 1536; idx += 128) {
    int q = idx >> 9, rem = idx & 511, rr = rem >> 5, mm = rem & 31;
    float s = lds_mat[0][q][rr][mm] + lds_mat[1][q][rr][mm]
            + lds_mat[2][q][rr][mm] + lds_mat[3][q][rr][mm];
    ws[WS1_OFF + base1 + idx] = s;
  }
  if (tid < 32) {
    int s01 = tid >> 4, rr = tid & 15;
    float s = lds_ns[0][s01][rr] + lds_ns[1][s01][rr]
            + lds_ns[2][s01][rr] + lds_ns[3][s01][rr];
    ws[WS2_OFF + (size_t)((b*NT_ + nt)*CS_ + cs)*32 + tid] = s;
    if (need_gsum) {
      float g = lds_gs[0][tid] + lds_gs[1][tid] + lds_gs[2][tid] + lds_gs[3][tid];
      ws[WS3_OFF + (size_t)(b*CS_ + cs)*32 + tid] = g;
    }
  }
}

// ---------------- Phase 2: finalize cost ------------------------------------
__global__ __launch_bounds__(256)
void cost_finalize_kernel(const float* __restrict__ objn,
                          const float* __restrict__ ws,
                          float* __restrict__ out) {
  const int b  = blockIdx.x / NT_;
  const int nt = blockIdx.x % NT_;
  for (int idx = threadIdx.x; idx < 512; idx += 256) {
    int r = idx >> 5, m = idx & 31;
    float P = 0.f, Ng = 0.f, S = 0.f, ns = 0.f, ss = 0.f, gs = 0.f;
    #pragma unroll
    for (int cs = 0; cs < CS_; cs++) {
      size_t base1 = (size_t)((b*NT_ + nt)*CS_ + cs) * 3 * 512;
      P  += ws[WS1_OFF + base1 + idx];
      Ng += ws[WS1_OFF + base1 + 512 + idx];
      S  += ws[WS1_OFF + base1 + 1024 + idx];
      size_t base2 = (size_t)((b*NT_ + nt)*CS_ + cs) * 32;
      ns += ws[WS2_OFF + base2 + r];
      ss += ws[WS2_OFF + base2 + 16 + r];
      gs += ws[WS3_OFF + (size_t)(b*CS_ + cs)*32 + m];
    }
    int n = nt*16 + r;
    float ox  = objn[b*N_ + n];
    float sob = __builtin_amdgcn_rcpf(1.f + __builtin_amdgcn_exp2f(ox * -LOG2E_F));
    float bce  = (P + ns - Ng) * (1.0f / (float)C_);
    float dice = 1.0f - (2.0f*S + 1.0f) * __builtin_amdgcn_rcpf(ss + gs + 1.0f);
    out[(size_t)(b*N_ + n)*M_ + m] = -2.0f*sob + 5.0f*bce + 5.0f*dice;
  }
}

// ---------------- Phase 3: warp-parallel Jonker–Volgenant LSAP --------------
__global__ __launch_bounds__(32)
void lsap_kernel(const float* __restrict__ cost, float* __restrict__ idx_out) {
  const int b   = blockIdx.x;
  const int tid = threadIdx.x;
  const float BIG = 1e30f;

  __shared__ float costT[M_][N_];
  __shared__ float u[M_], v[N_], shortest[N_];
  __shared__ int row4col[N_], col4row[M_], path[N_], SR[M_];
  __shared__ int SC[N_];
  __shared__ float s_minVal;
  __shared__ int s_i, s_sink;

  for (int k = tid; k < M_*N_; k += 32) {
    int m = k >> 7, n = k & 127;
    costT[m][n] = cost[(size_t)(b*N_ + n)*M_ + m];
  }
  #pragma unroll
  for (int q = 0; q < 4; q++) { int j = tid + 32*q; v[j] = 0.f; row4col[j] = -1; }
  u[tid] = 0.f; col4row[tid] = -1;
  __syncthreads();

  for (int cur = 0; cur < M_; cur++) {
    #pragma unroll
    for (int q = 0; q < 4; q++) {
      int j = tid + 32*q; shortest[j] = BIG; path[j] = -1; SC[j] = 0;
    }
    SR[tid] = 0;
    if (tid == 0) { s_i = cur; s_minVal = 0.f; s_sink = -1; }
    __syncthreads();

    while (true) {
      int   i  = s_i;
      float mv = s_minVal;
      if (tid == 0) SR[i] = 1;
      float ui = u[i];
      #pragma unroll
      for (int q = 0; q < 4; q++) {
        int j = tid + 32*q;
        if (!SC[j]) {
          float d = mv + costT[i][j] - ui - v[j];
          if (d < shortest[j]) { shortest[j] = d; path[j] = i; }
        }
      }
      __syncthreads();
      // argmin over masked shortest, first-index tiebreak (match jnp.argmin)
      float bv = BIG; int bj = N_;
      #pragma unroll
      for (int q = 0; q < 4; q++) {
        int j = tid + 32*q;
        float m2 = SC[j] ? BIG : shortest[j];
        if (m2 < bv || (m2 == bv && j < bj)) { bv = m2; bj = j; }
      }
      #pragma unroll
      for (int off = 16; off >= 1; off >>= 1) {
        float ov = __shfl_xor(bv, off);
        int   oj = __shfl_xor(bj, off);
        if (ov < bv || (ov == bv && oj < bj)) { bv = ov; bj = oj; }
      }
      if (tid == 0) {
        s_minVal = bv;
        SC[bj] = 1;
        int nxt = row4col[bj];
        if (nxt == -1) s_sink = bj; else s_i = nxt;
      }
      __syncthreads();
      if (s_sink >= 0) break;
    }

    float mv = s_minVal; int sink = s_sink;
    // dual updates (scipy rectangular_lsap)
    if (tid == cur)       u[tid] += mv;
    else if (SR[tid])     { int c4 = col4row[tid]; u[tid] += mv - shortest[c4]; }
    #pragma unroll
    for (int q = 0; q < 4; q++) {
      int j = tid + 32*q;
      if (SC[j]) v[j] -= mv - shortest[j];
    }
    __syncthreads();
    if (tid == 0) {
      int j = sink;
      while (true) {
        int i = path[j];
        row4col[j] = i;
        int t = col4row[i];
        col4row[i] = j;
        if (i == cur) break;
        j = t;
      }
    }
    __syncthreads();
  }
  idx_out[b*M_ + tid] = (float)col4row[tid];
}

// ---------------------------------------------------------------------------
extern "C" void kernel_launch(void* const* d_in, const int* in_sizes, int n_in,
                              void* d_out, int out_size, void* d_ws, size_t ws_size,
                              hipStream_t stream) {
  const float* objn = (const float*)d_in[0];   // [8,128]
  const float* pred = (const float*)d_in[1];   // [8,128,65536]
  const float* gt   = (const float*)d_in[2];   // [8,32,65536]
  float* out = (float*)d_out;                  // cost [8,128,32] then indices [8,32]
  float* ws  = (float*)d_ws;

  hipLaunchKernelGGL(cost_partial_kernel, dim3(B_*NT_*CS_), dim3(128), 0, stream,
                     pred, gt, ws);
  hipLaunchKernelGGL(cost_finalize_kernel, dim3(B_*NT_), dim3(256), 0, stream,
                     objn, ws, out);
  hipLaunchKernelGGL(lsap_kernel, dim3(B_), dim3(32), 0, stream,
                     out, out + (size_t)B_*N_*M_);
}
